// MultiHeadedSelfAttention_12077448037095
// MI455X (gfx1250) — compile-verified, matrix operand order checked by reading, not run on hardware
//
#include <hip/hip_runtime.h>
#include <hip/hip_bf16.h>

typedef unsigned short u16;
typedef unsigned int   u32;

typedef __attribute__((ext_vector_type(16))) __bf16 v16bf;
typedef __attribute__((ext_vector_type(8)))  float  v8f;
typedef __attribute__((ext_vector_type(4)))  int    v4i;

#if defined(__has_builtin)
#if __has_builtin(__builtin_amdgcn_global_load_async_to_lds_b128)
#define HAVE_ASYNC_LDS 1
#endif
#endif
#ifndef HAVE_ASYNC_LDS
#define HAVE_ASYNC_LDS 0
#endif

struct FragBits { uint4 lo; uint4 hi; };   // 32 bytes == 16 bf16

static __device__ __forceinline__ v16bf make_frag(const uint4& lo, const uint4& hi) {
  FragBits s{lo, hi};
  return __builtin_bit_cast(v16bf, s);
}

// WMMA 16-bit A/B fragment layout: lane L holds (row/col = L%16),
// K-chunks at koff=(L/16)*8 : elements [koff, koff+7] and [koff+16, koff+23].
// Caller passes p already offset by koff; chunks are uint4[0] and uint4[2].
static __device__ __forceinline__ v16bf load_frag16(const u16* p) {
  const uint4* u = reinterpret_cast<const uint4*>(p);
  return make_frag(u[0], u[2]);
}

static __device__ __forceinline__ u16 f2bf(float f) {
  u32 u = __builtin_bit_cast(u32, f);
  u += 0x7FFFu + ((u >> 16) & 1u);   // RNE
  return (u16)(u >> 16);
}

// 16-byte global -> LDS copy. Prefer the gfx1250 async-to-LDS path
// (ASYNCcnt-tracked DMA, no VGPR round trip); fall back to vector copy.
static __device__ __forceinline__ void copy16_g2l(u16* l, const u16* g) {
#if HAVE_ASYNC_LDS
  __builtin_amdgcn_global_load_async_to_lds_b128(
      (__attribute__((address_space(1))) v4i*)g,
      (__attribute__((address_space(3))) v4i*)l, 0, 0);
#else
  *reinterpret_cast<uint4*>(l) = *reinterpret_cast<const uint4*>(g);
#endif
}

static __device__ __forceinline__ void wait_async_copies() {
#if HAVE_ASYNC_LDS
#if defined(__has_builtin) && __has_builtin(__builtin_amdgcn_s_wait_asynccnt)
  __builtin_amdgcn_s_wait_asynccnt(0);
#else
  asm volatile("s_wait_asynccnt 0x0" ::: "memory");
#endif
#endif
}

constexpr int DIM = 1024;
constexpr int SEQ = 2048;
constexpr int BATCH = 2;
constexpr int HEADS = 16;
constexpr int HW = 64;        // head width

// ---------------------------------------------------------------------------
// Kernel 1: LayerNorm + convert to bf16. One 256-thread block per row.
// ---------------------------------------------------------------------------
__global__ __launch_bounds__(256) void ln_bf16_kernel(
    const float* __restrict__ x, const float* __restrict__ gamma,
    const float* __restrict__ beta, u16* __restrict__ xn) {
  const int row = blockIdx.x;          // 0 .. B*S-1
  const int tid = threadIdx.x;         // 256 threads * float4 = 1024
  const float4 v = reinterpret_cast<const float4*>(x + (size_t)row * DIM)[tid];
  float s  = v.x + v.y + v.z + v.w;
  float s2 = v.x * v.x + v.y * v.y + v.z * v.z + v.w * v.w;
  #pragma unroll
  for (int off = 16; off > 0; off >>= 1) {
    s  += __shfl_down(s,  off, 32);
    s2 += __shfl_down(s2, off, 32);
  }
  __shared__ float red0[8], red1[8];
  const int wid = tid >> 5, lid = tid & 31;
  if (lid == 0) { red0[wid] = s; red1[wid] = s2; }
  __syncthreads();
  float ts = 0.f, ts2 = 0.f;
  #pragma unroll
  for (int i = 0; i < 8; i++) { ts += red0[i]; ts2 += red1[i]; }
  const float mu   = ts * (1.0f / DIM);
  const float var  = ts2 * (1.0f / DIM) - mu * mu;
  const float rstd = rsqrtf(var + 1e-5f);
  const float4 g = reinterpret_cast<const float4*>(gamma)[tid];
  const float4 b = reinterpret_cast<const float4*>(beta)[tid];
  uint2 o;
  o.x = (u32)f2bf((v.x - mu) * rstd * g.x + b.x) |
        ((u32)f2bf((v.y - mu) * rstd * g.y + b.y) << 16);
  o.y = (u32)f2bf((v.z - mu) * rstd * g.z + b.z) |
        ((u32)f2bf((v.w - mu) * rstd * g.w + b.w) << 16);
  reinterpret_cast<uint2*>(xn + (size_t)row * DIM)[tid] = o;
}

// ---------------------------------------------------------------------------
// Kernel 2: transpose + bf16-convert weights: Wt[j][k] = (bf16) W[k][j]
// ---------------------------------------------------------------------------
__global__ __launch_bounds__(256) void wt_bf16_kernel(
    const float* __restrict__ wq, const float* __restrict__ wk,
    const float* __restrict__ wv, u16* __restrict__ wt) {
  __shared__ float tile[32][33];
  const float* w = (blockIdx.z == 0) ? wq : (blockIdx.z == 1) ? wk : wv;
  u16* o = wt + (size_t)blockIdx.z * DIM * DIM;
  const int tx = threadIdx.x, ty = threadIdx.y;        // 32 x 8
  #pragma unroll
  for (int i = 0; i < 4; i++) {
    const int k = blockIdx.y * 32 + ty + i * 8;
    const int j = blockIdx.x * 32 + tx;
    tile[ty + i * 8][tx] = w[(size_t)k * DIM + j];     // tile[kloc][jloc]
  }
  __syncthreads();
  #pragma unroll
  for (int i = 0; i < 4; i++) {
    const int j = blockIdx.x * 32 + ty + i * 8;
    const int k = blockIdx.y * 32 + tx;
    o[(size_t)j * DIM + k] = f2bf(tile[tx][ty + i * 8]);
  }
}

// ---------------------------------------------------------------------------
// Kernel 3: QKV GEMM on WMMA bf16. grid=(32 Mgroups, 16 Ngroups, 3 proj),
// 256 threads = 8 waves; wave handles a 16x64 tile over K=1024 (32 k-steps).
// q,k stored (B,H,S,W); v stored transposed (B,H,W,S). Scale 1/8 folded in q.
// ---------------------------------------------------------------------------
__global__ __launch_bounds__(256) void qkv_wmma_kernel(
    const u16* __restrict__ xn, const u16* __restrict__ wt,
    const float* __restrict__ bq, const float* __restrict__ bk,
    const float* __restrict__ bv, u16* __restrict__ qO,
    u16* __restrict__ kO, u16* __restrict__ vtO) {
  const int proj = blockIdx.z;
  const int wave = threadIdx.x >> 5;
  const int lane = threadIdx.x & 31;
  const int mtile = blockIdx.x * 8 + wave;           // 0..255, rows mtile*16
  const int ncol0 = blockIdx.y * 64;
  const u16* wsel = wt + (size_t)proj * DIM * DIM;
  const float* bias = (proj == 0) ? bq : (proj == 1) ? bk : bv;

  const int lq = lane & 15;
  const int koff = (lane >> 4) * 8;

  v8f acc[4] = {};
  const u16* arow = xn + (size_t)(mtile * 16 + lq) * DIM;
  for (int kb = 0; kb < DIM; kb += 32) {
    const v16bf a = load_frag16(arow + kb + koff);
    #pragma unroll
    for (int nt = 0; nt < 4; nt++) {
      const v16bf bfr =
          load_frag16(wsel + (size_t)(ncol0 + nt * 16 + lq) * DIM + kb + koff);
      acc[nt] = __builtin_amdgcn_wmma_f32_16x16x32_bf16(
          false, a, false, bfr, (short)0, acc[nt], false, false);
    }
  }

  const int half = lane >> 4;
  #pragma unroll
  for (int nt = 0; nt < 4; nt++) {
    const int col = ncol0 + nt * 16 + lq;            // N = lane%16
    const float bvl = bias[col];
    const int h = col >> 6, w = col & 63;
    #pragma unroll
    for (int r = 0; r < 8; r++) {
      const int m = r + half * 8;                    // M = vgpr + 8*(lane/16)
      const int row = mtile * 16 + m;
      const int b = row >> 11, s = row & (SEQ - 1);
      float val = acc[nt][r] + bvl;
      if (proj == 0) {
        val *= 0.125f;                               // 1/sqrt(64)
        qO[((size_t)(b * HEADS + h) * SEQ + s) * HW + w] = f2bf(val);
      } else if (proj == 1) {
        kO[((size_t)(b * HEADS + h) * SEQ + s) * HW + w] = f2bf(val);
      } else {
        vtO[((size_t)(b * HEADS + h) * HW + w) * SEQ + s] = f2bf(val);
      }
    }
  }
}

// ---------------------------------------------------------------------------
// Kernel 4: flash attention. 128 threads = 4 waves (same b,h), one 16-query
// tile per wave. Per key tile (32 keys):
//   * block cooperatively stages K(32x64) and V^T(64x32) into LDS via
//     gfx1250 async-to-LDS DMA (ASYNCcnt), shared by all 4 waves
//   * scores^T = K x q^T on WMMA -> online softmax (per-lane + shfl_xor 16)
//   * P -> LDS bf16 (wave-local, s_wait_dscnt) -> B-fragment
//   * h^T += V^T x P^T on WMMA
// ---------------------------------------------------------------------------
__global__ __launch_bounds__(128) void attn_kernel(
    const u16* __restrict__ q, const u16* __restrict__ k,
    const u16* __restrict__ vt, const int* __restrict__ mask,
    const float* __restrict__ x, float* __restrict__ out) {
  __shared__ float pen[SEQ];                          // additive mask penalty
  __shared__ __align__(16) u16 KtL[32 * 64];          // K tile   (keys x W)
  __shared__ __align__(16) u16 VtL[64 * 32];          // V^T tile (W x keys)
  __shared__ __align__(16) u16 Pt[4][16 * 32];        // per-wave P tile

  const int tid = threadIdx.x, wave = tid >> 5, lane = tid & 31;
  const int bh = blockIdx.x >> 5;                     // b*H + h
  const int qgrp = blockIdx.x & 31;
  const int b = bh >> 4, h = bh & 15;
  const int qbase = (qgrp * 4 + wave) * 16;

  for (int i = tid; i < SEQ; i += 128)
    pen[i] = -10000.0f * (1.0f - (float)mask[b * SEQ + i]);

  const u16* qbh = q + (size_t)bh * SEQ * HW;
  const u16* kbh = k + (size_t)bh * SEQ * HW;
  const u16* vbh = vt + (size_t)bh * HW * SEQ;

  const int lq = lane & 15, half = lane >> 4, koff = half * 8;

  // q^T B-fragments for the two 32-wide K-steps of the 64-dim head
  v16bf qf[2];
  {
    const u16* qr = qbh + (size_t)(qbase + lq) * HW;
    qf[0] = load_frag16(qr + 0 + koff);
    qf[1] = load_frag16(qr + 32 + koff);
  }

  v8f o[4] = {};
  float m_i = -3.0e38f, l_i = 0.0f;
  u16* Pl = Pt[wave];

  for (int kb = 0; kb < SEQ; kb += 32) {
    // ---- cooperative staging of K / V^T tiles (8 KB) through async DMA ----
    __syncthreads();                                  // prev tile fully read
    {
      const u16* gk = kbh + (size_t)kb * HW;          // 4 KB, contiguous
      #pragma unroll
      for (int c = 0; c < 2; c++) {
        const int chunk = tid + c * 128;              // 256 x 16B
        copy16_g2l(KtL + chunk * 8, gk + chunk * 8);
      }
      #pragma unroll
      for (int c = 0; c < 2; c++) {
        const int chunk = tid + c * 128;              // 64 rows x 4 chunks
        const int w = chunk >> 2, off = (chunk & 3) * 8;
        copy16_g2l(VtL + w * 32 + off, vbh + (size_t)w * SEQ + kb + off);
      }
    }
    if (kb + 32 < SEQ) {
      __builtin_prefetch(kbh + (size_t)(kb + 32) * HW + tid * 8, 0, 1);
      __builtin_prefetch(vbh + (size_t)(tid >> 1) * SEQ + kb + 32, 0, 1);
    }
    wait_async_copies();
    __syncthreads();                                  // tiles visible to block

    // ---- scores^T tile: 32 keys x 16 queries, two 16x16 accumulators ----
    v8f s0 = {}, s1 = {};
    #pragma unroll
    for (int ks = 0; ks < 2; ks++) {
      const v16bf a0 = load_frag16(KtL + (size_t)lq * HW + ks * 32 + koff);
      s0 = __builtin_amdgcn_wmma_f32_16x16x32_bf16(
          false, a0, false, qf[ks], (short)0, s0, false, false);
      const v16bf a1 =
          load_frag16(KtL + (size_t)(16 + lq) * HW + ks * 32 + koff);
      s1 = __builtin_amdgcn_wmma_f32_16x16x32_bf16(
          false, a1, false, qf[ks], (short)0, s1, false, false);
    }
    // additive mask (keys live along vgpr index + lane half)
    #pragma unroll
    for (int r = 0; r < 8; r++) {
      s0[r] += pen[kb + half * 8 + r];
      s1[r] += pen[kb + 16 + half * 8 + r];
    }
    // ---- online softmax over keys: per-lane 16 values + partner (xor 16) --
    float mloc = -3.0e38f;
    #pragma unroll
    for (int r = 0; r < 8; r++) mloc = fmaxf(mloc, fmaxf(s0[r], s1[r]));
    mloc = fmaxf(mloc, __shfl_xor(mloc, 16, 32));
    const float mnew = fmaxf(m_i, mloc);
    const float corr = __expf(m_i - mnew);
    float p0[8], p1[8], psum = 0.0f;
    #pragma unroll
    for (int r = 0; r < 8; r++) {
      p0[r] = __expf(s0[r] - mnew);
      p1[r] = __expf(s1[r] - mnew);
      psum += p0[r] + p1[r];
    }
    psum += __shfl_xor(psum, 16, 32);
    l_i = l_i * corr + psum;
    m_i = mnew;
    #pragma unroll
    for (int t = 0; t < 4; t++)
      #pragma unroll
      for (int r = 0; r < 8; r++) o[t][r] *= corr;

    // ---- P -> LDS, query-major rows of 32 keys (bf16, packed pairs) ------
    u32* Pw = reinterpret_cast<u32*>(Pl);
    #pragma unroll
    for (int r = 0; r < 8; r += 2) {
      Pw[(lq * 32 + half * 8 + r) >> 1] =
          (u32)f2bf(p0[r]) | ((u32)f2bf(p0[r + 1]) << 16);
      Pw[(lq * 32 + 16 + half * 8 + r) >> 1] =
          (u32)f2bf(p1[r]) | ((u32)f2bf(p1[r + 1]) << 16);
    }
    asm volatile("s_wait_dscnt 0x0" ::: "memory");    // wave-local LDS RAW

    const v16bf pfrag = load_frag16(Pl + lq * 32 + koff);   // B = P^T
    #pragma unroll
    for (int t = 0; t < 4; t++) {                     // h^T rows w = t*16..+15
      const v16bf vfrag = load_frag16(VtL + (size_t)(t * 16 + lq) * 32 + koff);
      o[t] = __builtin_amdgcn_wmma_f32_16x16x32_bf16(
          false, vfrag, false, pfrag, (short)0, o[t], false, false);
    }
  }

  // epilogue: divide by softmax sum, add residual, store fp32
  const float inv = 1.0f / (l_i + 1e-20f);
  const int s = qbase + lq;
  #pragma unroll
  for (int t = 0; t < 4; t++) {
    #pragma unroll
    for (int r = 0; r < 8; r++) {
      const int w = t * 16 + r + 8 * half;
      const size_t oidx = ((size_t)b * SEQ + s) * DIM + h * HW + w;
      out[oidx] = o[t][r] * inv + x[oidx];
    }
  }
}

// ---------------------------------------------------------------------------
extern "C" void kernel_launch(void* const* d_in, const int* in_sizes, int n_in,
                              void* d_out, int out_size, void* d_ws,
                              size_t ws_size, hipStream_t stream) {
  const float* x     = (const float*)d_in[0];
  const int*   mask  = (const int*)d_in[1];
  const float* gamma = (const float*)d_in[2];
  const float* beta  = (const float*)d_in[3];
  const float* wq    = (const float*)d_in[4];
  const float* bq    = (const float*)d_in[5];
  const float* wk    = (const float*)d_in[6];
  const float* bk    = (const float*)d_in[7];
  const float* wv    = (const float*)d_in[8];
  const float* bv    = (const float*)d_in[9];
  float* out = (float*)d_out;

  char* ws = (char*)d_ws;
  const size_t MB = 1024 * 1024;
  u16* xn = (u16*)ws;                         // 8 MB  (4096x1024 bf16)
  u16* wt = (u16*)(ws + 8 * MB);              // 6 MB  (3x 1024x1024 bf16, transposed)
  u16* qb = (u16*)(ws + 14 * MB);             // 8 MB  (B,H,S,W)
  u16* kb = (u16*)(ws + 22 * MB);             // 8 MB  (B,H,S,W)
  u16* vt = (u16*)(ws + 30 * MB);             // 8 MB  (B,H,W,S)

  ln_bf16_kernel<<<BATCH * SEQ, 256, 0, stream>>>(x, gamma, beta, xn);

  dim3 tg(32, 32, 3), tb(32, 8, 1);
  wt_bf16_kernel<<<tg, tb, 0, stream>>>(wq, wk, wv, wt);

  dim3 gg(32, 16, 3);
  qkv_wmma_kernel<<<gg, 256, 0, stream>>>(xn, wt, bq, bk, bv, qb, kb, vt);

  attn_kernel<<<BATCH * HEADS * (SEQ / 16) / 4, 128, 0, stream>>>(
      qb, kb, vt, mask, x, out);
}
